// GCN_71451075936470
// MI455X (gfx1250) — matrix operand order repaired and checked by reading, not compiled, not run on hardware
//
#include <hip/hip_runtime.h>
#include <hip/hip_bf16.h>

#define NN 50000
#define EE 800000
#define GG 64
#define BN_EPS 1e-5f

typedef __bf16  v16bf __attribute__((ext_vector_type(16)));
typedef __bf16  v8bf  __attribute__((ext_vector_type(8)));
typedef __bf16  v4bf  __attribute__((ext_vector_type(4)));
typedef float   v8f   __attribute__((ext_vector_type(8)));

// ---------------------------------------------------------------------------
// Degree / normalization (computed once, reused by all 4 layers)
// ---------------------------------------------------------------------------
__global__ void deg_init_kernel(float* deg) {
    int i = blockIdx.x * blockDim.x + threadIdx.x;
    if (i < NN) deg[i] = 1.0f;               // self-loop weight = 1.0
}

__global__ void deg_acc_kernel(const int* __restrict__ dst,
                               const float* __restrict__ w,
                               float* __restrict__ deg) {
    int e = blockIdx.x * blockDim.x + threadIdx.x;
    if (e < EE) atomicAdd(&deg[dst[e]], w[e]);
}

__global__ void deg_rsqrt_kernel(const float* __restrict__ deg,
                                 float* __restrict__ dis,
                                 float* __restrict__ selfn) {
    int i = blockIdx.x * blockDim.x + threadIdx.x;
    if (i < NN) {
        float d  = deg[i];
        float r  = d > 0.0f ? rsqrtf(d) : 0.0f;
        dis[i]   = r;
        selfn[i] = r * r;                    // self-loop norm dis^2 * 1.0
    }
}

__global__ void norm_kernel(const int* __restrict__ src,
                            const int* __restrict__ dst,
                            const float* __restrict__ w,
                            const float* __restrict__ dis,
                            float* __restrict__ norm) {
    int e = blockIdx.x * blockDim.x + threadIdx.x;
    if (e < EE) norm[e] = dis[src[e]] * w[e] * dis[dst[e]];
}

// ---------------------------------------------------------------------------
// fp32 -> bf16 conversion (activations, once per layer) and
// weight transpose+convert: Wt[n*FIN + k] = (bf16) W[k*FOUT + n]
// ---------------------------------------------------------------------------
__global__ void cvt_bf16_kernel(const float* __restrict__ in,
                                __bf16* __restrict__ out, int n4) {
    int idx = blockIdx.x * blockDim.x + threadIdx.x;
    if (idx >= n4) return;
    float4 v = ((const float4*)in)[idx];
    v4bf o = {(__bf16)v.x, (__bf16)v.y, (__bf16)v.z, (__bf16)v.w};
    ((v4bf*)out)[idx] = o;
}

template <int FIN, int FOUT>
__global__ void wt_bf16_kernel(const float* __restrict__ W,
                               __bf16* __restrict__ Wt) {
    int idx = blockIdx.x * blockDim.x + threadIdx.x;
    if (idx >= FIN * FOUT) return;
    int n = idx / FIN;
    int k = idx % FIN;
    Wt[idx] = (__bf16)W[k * FOUT + n];
}

// ---------------------------------------------------------------------------
// GEMM: Out[M,FOUT] = A[M,FIN] * W[FIN,FOUT], bf16 WMMA, fp32 accumulate.
// A: bf16 row-major [M,FIN].  Wt: bf16 [FOUT,FIN] (pre-transposed).
// No LDS: WMMA fragments are contiguous 16B runs in memory.
//   A frag (16-bit A 16x32): lane row = lane&15; K base = (lane>>4)*8;
//                            elems {base..base+7, base+16..base+23}
//   B frag (16-bit B 32x16): lane col = lane&15; K base = (lane>>4)*8; elems as A
// One wave computes a 16x(16*NB) output block (NB accumulators, A reuse).
// ---------------------------------------------------------------------------
template <int FIN, int FOUT, int NB>
__global__ void gemm_wmma_kernel(const __bf16* __restrict__ A,
                                 const __bf16* __restrict__ Wt,
                                 float* __restrict__ Out) {
    constexpr int MT = NN / 16;                 // 3125 m-tiles
    constexpr int NBLK = FOUT / (16 * NB);      // n-blocks
    constexpr int T = MT * NBLK;                // wave-tiles total

    const int tid  = threadIdx.x;
    const int lane = tid & 31;
    const int wave = tid >> 5;
    const int tile = blockIdx.x * 4 + wave;     // 4 waves / block
    if (tile >= T) return;                      // no barriers below: safe

    const int mt = tile % MT;
    const int nb0 = tile / MT;
    const int m0 = mt * 16;
    const int n0 = nb0 * (16 * NB);
    const int half = lane >> 4;
    const int mr   = lane & 15;

    const __bf16* arow = A + (m0 + mr) * FIN;
    const __bf16* brow = Wt + (n0 + mr) * FIN;

    v8f acc[NB];
    #pragma unroll
    for (int nb = 0; nb < NB; ++nb)
        acc[nb] = (v8f){0.f, 0.f, 0.f, 0.f, 0.f, 0.f, 0.f, 0.f};

    #pragma unroll
    for (int k0 = 0; k0 < FIN; k0 += 32) {
        v8bf alo = *(const v8bf*)(arow + k0 + half * 8);
        v8bf ahi = *(const v8bf*)(arow + k0 + half * 8 + 16);
        v16bf a = __builtin_shufflevector(alo, ahi,
                    0, 1, 2, 3, 4, 5, 6, 7, 8, 9, 10, 11, 12, 13, 14, 15);
        #pragma unroll
        for (int nb = 0; nb < NB; ++nb) {
            const __bf16* bp = brow + nb * 16 * FIN + k0 + half * 8;
            v8bf blo = *(const v8bf*)(bp);
            v8bf bhi = *(const v8bf*)(bp + 16);
            v16bf b = __builtin_shufflevector(blo, bhi,
                        0, 1, 2, 3, 4, 5, 6, 7, 8, 9, 10, 11, 12, 13, 14, 15);
            acc[nb] = __builtin_amdgcn_wmma_f32_16x16x32_bf16(
                          false, a, false, b, (short)0, acc[nb], false, false);
        }
    }

    // C/D layout: VGPR v, lanes 0-15 -> M=v; lanes 16-31 -> M=v+8; N=lane&15.
    const int mbase = half * 8;
    #pragma unroll
    for (int nb = 0; nb < NB; ++nb)
        #pragma unroll
        for (int v = 0; v < 8; ++v)
            Out[(m0 + mbase + v) * FOUT + n0 + nb * 16 + mr] = acc[nb][v];
}

// ---------------------------------------------------------------------------
// Scatter (GCN propagation), float4-vectorized.
// ---------------------------------------------------------------------------
template <int F>
__global__ void scatter_init_kernel(const float* __restrict__ xw,
                                    const float* __restrict__ selfn,
                                    const float* __restrict__ bias,
                                    float* __restrict__ out) {
    constexpr int F4 = F / 4;
    int idx = blockIdx.x * blockDim.x + threadIdx.x;
    if (idx >= NN * F4) return;
    int i = idx / F4;
    int q = idx % F4;
    float  s  = selfn[i];
    float4 xv = ((const float4*)xw)[idx];
    float4 bv = ((const float4*)bias)[q];
    float4 o;
    o.x = s * xv.x + bv.x;
    o.y = s * xv.y + bv.y;
    o.z = s * xv.z + bv.z;
    o.w = s * xv.w + bv.w;
    ((float4*)out)[idx] = o;
}

template <int F>
__global__ void scatter_edges_kernel(const float* __restrict__ xw,
                                     const float* __restrict__ norm,
                                     const int* __restrict__ src,
                                     const int* __restrict__ dst,
                                     float* __restrict__ out) {
    constexpr int F4 = F / 4;
    unsigned int idx = blockIdx.x * blockDim.x + threadIdx.x;
    if (idx >= (unsigned int)EE * F4) return;
    int e = idx / F4;
    int q = idx % F4;
    int s = src[e];
    int d = dst[e];
    float  nv = norm[e];
    float4 xv = ((const float4*)xw)[s * F4 + q];
    float* o  = &out[d * F + q * 4];
    atomicAdd(o + 0, nv * xv.x);
    atomicAdd(o + 1, nv * xv.y);
    atomicAdd(o + 2, nv * xv.z);
    atomicAdd(o + 3, nv * xv.w);
}

// ---------------------------------------------------------------------------
// BatchNorm (batch stats, biased var) + ReLU: zero -> stats -> apply.
// ---------------------------------------------------------------------------
__global__ void zero2_kernel(float* __restrict__ a, float* __restrict__ b, int n) {
    int i = blockIdx.x * blockDim.x + threadIdx.x;
    if (i < n) { a[i] = 0.f; b[i] = 0.f; }
}

template <int F>
__global__ void bn_stats_kernel(const float* __restrict__ h,
                                float* __restrict__ colsum,
                                float* __restrict__ colsq) {
    constexpr int LANES = 256 / F;      // row-groups per block
    constexpr int RPL   = 200 / LANES;  // rows per row-group (NN = 250*200)
    __shared__ float ss[256];
    __shared__ float sq[256];
    const int t    = threadIdx.x;
    const int f    = t % F;
    const int grp  = t / F;
    const int row0 = blockIdx.x * 200 + grp * RPL;

    float sum = 0.f, sqv = 0.f;
    for (int r = 0; r < RPL; ++r) {
        float v = h[(row0 + r) * F + f];   // coalesced: consecutive t -> consecutive f
        sum += v;
        sqv += v * v;
    }
    ss[t] = sum;
    sq[t] = sqv;
    __syncthreads();
    #pragma unroll
    for (int off = 128; off >= F; off >>= 1) {
        if (t < off) { ss[t] += ss[t + off]; sq[t] += sq[t + off]; }
        __syncthreads();
    }
    if (t < F) {
        atomicAdd(&colsum[f], ss[t]);
        atomicAdd(&colsq[f],  sq[t]);
    }
}

template <int F>
__global__ void bn_apply_kernel(float* __restrict__ h,
                                const float* __restrict__ colsum,
                                const float* __restrict__ colsq,
                                const float* __restrict__ g,
                                const float* __restrict__ be) {
    constexpr int F4 = F / 4;
    int idx = blockIdx.x * blockDim.x + threadIdx.x;
    if (idx >= NN * F4) return;
    int q = idx % F4;
    int f = q * 4;
    const float inv_n = 1.0f / (float)NN;
    float4 v = ((float4*)h)[idx];
    float* vp = &v.x;
    #pragma unroll
    for (int j = 0; j < 4; ++j) {
        int   cidx = f + j;
        float mu  = colsum[cidx] * inv_n;
        float var = colsq[cidx] * inv_n - mu * mu;
        float sc  = rsqrtf(var + BN_EPS) * g[cidx];
        float r   = (vp[j] - mu) * sc + be[cidx];
        vp[j] = r > 0.f ? r : 0.f;
    }
    ((float4*)h)[idx] = v;
}

// ---------------------------------------------------------------------------
// Global mean pool + linear head + sigmoid
// ---------------------------------------------------------------------------
__global__ void pool_zero_kernel(float* __restrict__ pooled,
                                 float* __restrict__ cnt) {
    int i = blockIdx.x * blockDim.x + threadIdx.x;
    if (i < GG * 32) pooled[i] = 0.f;
    if (i < GG) cnt[i] = 0.f;
}

__global__ void pool_acc_kernel(const float* __restrict__ h,
                                const int* __restrict__ batch,
                                float* __restrict__ pooled,
                                float* __restrict__ cnt) {
    int idx = blockIdx.x * blockDim.x + threadIdx.x;   // over NN*32
    if (idx >= NN * 32) return;
    int i = idx >> 5;
    int f = idx & 31;
    int gidx = batch[i];
    atomicAdd(&pooled[gidx * 32 + f], h[idx]);
    if (f == 0) atomicAdd(&cnt[gidx], 1.0f);
}

__global__ void head_kernel(const float* __restrict__ pooled,
                            const float* __restrict__ cnt,
                            const float* __restrict__ Wl,
                            const float* __restrict__ bl,
                            float* __restrict__ out) {
    int gidx = blockIdx.x * blockDim.x + threadIdx.x;
    if (gidx >= GG) return;
    float c = fmaxf(cnt[gidx], 1.0f);
    float acc = bl[0];
    #pragma unroll
    for (int f = 0; f < 32; ++f)
        acc += (pooled[gidx * 32 + f] / c) * Wl[f];
    out[gidx] = 1.0f / (1.0f + expf(-acc));
}

// ---------------------------------------------------------------------------
// Host-side orchestration
// ---------------------------------------------------------------------------
template <int FIN, int FOUT, int NB>
static void launch_layer(const float* hin, const float* W, const float* b,
                         const float* g, const float* be,
                         __bf16* hbf, __bf16* wtbf,
                         float* xw, float* hout,
                         const int* src, const int* dst,
                         const float* selfn, const float* norm,
                         float* colsum, float* colsq, hipStream_t stream) {
    // pre-pass: activations -> bf16, weights -> transposed bf16
    int n4 = NN * FIN / 4;
    cvt_bf16_kernel<<<(n4 + 255) / 256, 256, 0, stream>>>(hin, hbf, n4);
    wt_bf16_kernel<FIN, FOUT><<<(FIN * FOUT + 255) / 256, 256, 0, stream>>>(W, wtbf);

    // GEMM: 4 waves per block, flat wave-tile index
    constexpr int T = (NN / 16) * (FOUT / (16 * NB));
    gemm_wmma_kernel<FIN, FOUT, NB><<<(T + 3) / 4, 128, 0, stream>>>(hbf, wtbf, xw);

    constexpr int F4 = FOUT / 4;
    scatter_init_kernel<FOUT><<<(NN * F4 + 255) / 256, 256, 0, stream>>>(
        xw, selfn, b, hout);
    scatter_edges_kernel<FOUT>
        <<<((unsigned)EE * F4 + 255u) / 256u, 256, 0, stream>>>(
            xw, norm, src, dst, hout);

    zero2_kernel<<<1, 256, 0, stream>>>(colsum, colsq, FOUT);
    bn_stats_kernel<FOUT><<<NN / 200, 256, 0, stream>>>(hout, colsum, colsq);
    bn_apply_kernel<FOUT><<<(NN * F4 + 255) / 256, 256, 0, stream>>>(
        hout, colsum, colsq, g, be);
}

extern "C" void kernel_launch(void* const* d_in, const int* in_sizes, int n_in,
                              void* d_out, int out_size, void* d_ws, size_t ws_size,
                              hipStream_t stream) {
    const float* x   = (const float*)d_in[0];
    const int*   ei  = (const int*)d_in[1];   // [2,E] int32
    const float* ew  = (const float*)d_in[2];
    const int*   bat = (const int*)d_in[3];
    const float* W1 = (const float*)d_in[4],  *b1 = (const float*)d_in[5];
    const float* g1 = (const float*)d_in[6],  *be1 = (const float*)d_in[7];
    const float* W2 = (const float*)d_in[8],  *b2 = (const float*)d_in[9];
    const float* g2 = (const float*)d_in[10], *be2 = (const float*)d_in[11];
    const float* W3 = (const float*)d_in[12], *b3 = (const float*)d_in[13];
    const float* g3 = (const float*)d_in[14], *be3 = (const float*)d_in[15];
    const float* W4 = (const float*)d_in[16], *b4 = (const float*)d_in[17];
    const float* g4 = (const float*)d_in[18], *be4 = (const float*)d_in[19];
    const float* Wl = (const float*)d_in[20], *bl = (const float*)d_in[21];

    const int* src = ei;          // row 0
    const int* dst = ei + EE;     // row 1

    // workspace carve-up (floats); every block offset stays 16B-aligned
    float* ws     = (float*)d_ws;
    float* deg    = ws;                          // N
    float* dis    = deg + NN;                    // N
    float* selfn  = dis + NN;                    // N
    float* norm   = selfn + NN;                  // E
    float* colsum = norm + EE;                   // 256
    float* colsq  = colsum + 256;                // 256
    float* pooled = colsq + 256;                 // G*32
    float* cnt    = pooled + GG * 32;            // G
    float* xw     = cnt + GG;                    // N*256 (max Fout)
    float* hA     = xw + (size_t)NN * 256;       // N*256
    float* hB     = hA + (size_t)NN * 256;       // N*256
    __bf16* hbf   = (__bf16*)(hB + (size_t)NN * 256);   // N*256 bf16
    __bf16* wtbf  = hbf + (size_t)NN * 256;             // 256*256 bf16

    // degree + normalization coefficients (once, reused by all layers)
    deg_init_kernel<<<(NN + 255) / 256, 256, 0, stream>>>(deg);
    deg_acc_kernel<<<(EE + 255) / 256, 256, 0, stream>>>(dst, ew, deg);
    deg_rsqrt_kernel<<<(NN + 255) / 256, 256, 0, stream>>>(deg, dis, selfn);
    norm_kernel<<<(EE + 255) / 256, 256, 0, stream>>>(src, dst, ew, dis, norm);

    // layers: 128 -> 256 -> 128 -> 64 -> 32
    launch_layer<128, 256, 4>(x,  W1, b1, g1, be1, hbf, wtbf, xw, hA, src, dst, selfn, norm, colsum, colsq, stream);
    launch_layer<256, 128, 4>(hA, W2, b2, g2, be2, hbf, wtbf, xw, hB, src, dst, selfn, norm, colsum, colsq, stream);
    launch_layer<128, 64,  4>(hB, W3, b3, g3, be3, hbf, wtbf, xw, hA, src, dst, selfn, norm, colsum, colsq, stream);
    launch_layer<64,  32,  2>(hA, W4, b4, g4, be4, hbf, wtbf, xw, hB, src, dst, selfn, norm, colsum, colsq, stream);

    // pooling + head
    pool_zero_kernel<<<(GG * 32 + 255) / 256, 256, 0, stream>>>(pooled, cnt);
    pool_acc_kernel<<<(NN * 32 + 255) / 256, 256, 0, stream>>>(hB, bat, pooled, cnt);
    head_kernel<<<1, 64, 0, stream>>>(pooled, cnt, Wl, bl, (float*)d_out);
}
